// MllamaTextCrossAttention_33088428049112
// MI455X (gfx1250) — matrix operand order repaired and checked by reading, not compiled, not run on hardware
//
#include <hip/hip_runtime.h>

typedef __attribute__((ext_vector_type(16))) __bf16 bf16x16;
typedef __attribute__((ext_vector_type(8)))  __bf16 bf16x8;
typedef __attribute__((ext_vector_type(8)))  float  f32x8;
typedef __attribute__((ext_vector_type(4)))  unsigned int u32x4;
typedef __attribute__((ext_vector_type(8)))  int i32x8;
typedef __attribute__((ext_vector_type(4)))  int i32x4;

static __device__ __forceinline__ bf16x16 cat16(bf16x8 lo, bf16x8 hi) {
    return __builtin_shufflevector(lo, hi, 0,1,2,3,4,5,6,7,8,9,10,11,12,13,14,15);
}

static __device__ __forceinline__ f32x8 wmma_bf16(bf16x16 a, bf16x16 b, f32x8 c) {
    // D = A(16x32 bf16) * B(32x16 bf16) + C(16x16 f32)
    return __builtin_amdgcn_wmma_f32_16x16x32_bf16(false, a, false, b, (short)0, c, false, false);
}

// Issue one TDM 2D tile load: tile_d0 x tile_d1 elements (2B each), row-major,
// global row stride 'stride0' elements, valid extents tdim0/tdim1 (OOB -> zero),
// packed contiguously at LDS byte address 'lds_addr'.
static __device__ __forceinline__ void tdm_load_2d_bf16(unsigned int lds_addr,
                                                        const __bf16* gptr,
                                                        unsigned int tdim0, unsigned int tdim1,
                                                        unsigned int tile_d0, unsigned int tile_d1,
                                                        unsigned long long stride0) {
    unsigned long long ga = (unsigned long long)(uintptr_t)gptr;
    u32x4 g0;
    g0.x = 1u;                                        // count=1, user mode
    g0.y = lds_addr;                                  // lds_addr [63:32]
    g0.z = (unsigned int)ga;                          // global_addr low
    g0.w = (unsigned int)(ga >> 32) | (2u << 30);     // global_addr high | type=2
    i32x8 g1;
    g1[0] = (int)(1u << 16);                          // wg_mask=0, data_size=1 (2B)
    g1[1] = (int)((tdim0 & 0xFFFFu) << 16);           // tensor_dim0[15:0] @ bits[63:48]
    g1[2] = (int)((tdim0 >> 16) | ((tdim1 & 0xFFFFu) << 16));  // td0 hi, td1 lo
    g1[3] = (int)((tdim1 >> 16) | (tile_d0 << 16));   // td1 hi, tile_dim0
    g1[4] = (int)(tile_d1 & 0xFFFFu);                 // tile_dim1, tile_dim2=0
    g1[5] = (int)(unsigned int)(stride0 & 0xFFFFFFFFull);      // dim0_stride low32
    g1[6] = (int)(unsigned int)(stride0 >> 32);       // dim0_stride hi16, dim1_stride lo16=0
    g1[7] = 0;
    i32x4 gz4 = {0, 0, 0, 0};
    i32x8 gz8 = {0, 0, 0, 0, 0, 0, 0, 0};
    __builtin_amdgcn_tensor_load_to_lds(g0, g1, gz4, gz4, gz8, 0);
}

// ---------------------------------------------------------------------------
// GEMM: C[M x N] (f32) = A[M x K] * B[N x K]^T ; A is f32 or bf16, B is f32.
// Tiles: BM=128, BN=128, BK=32. 256 threads = 8 waves, wave -> 32x64 subtile.
// Register double-buffer: prefetch tile k+1 while computing tile k.
// ---------------------------------------------------------------------------
template <bool ABF16>
__global__ __launch_bounds__(256) void gemm_nt_kernel(const void* __restrict__ Av,
                                                      const float* __restrict__ B,
                                                      float* __restrict__ C,
                                                      int M, int N, int K) {
    __shared__ __align__(16) __bf16 As[128 * 32];
    __shared__ __align__(16) __bf16 Bs[128 * 32];

    const int t    = threadIdx.x;
    const int m0   = blockIdx.y * 128;
    const int n0   = blockIdx.x * 128;
    const int wave = t >> 5, lane = t & 31;
    const int hf   = lane >> 4, lq = lane & 15;
    const int wm   = wave & 3, wn = wave >> 2;

    const float*  Af = (const float*)Av;
    const __bf16* Ab = (const __bf16*)Av;

    f32x8 acc[2][4];
#pragma unroll
    for (int i = 0; i < 2; i++)
#pragma unroll
        for (int j = 0; j < 4; j++) acc[i][j] = (f32x8){0.f,0.f,0.f,0.f,0.f,0.f,0.f,0.f};

    union pack4 { uint2 u; __bf16 h[4]; };
    pack4 pa[4], pb[4];
    const int c = (t & 7) * 4;

    // ---- load tile kb into registers (converted to packed bf16) ----
    auto load_tile = [&](int kb, pack4* ra, pack4* rb) {
#pragma unroll
        for (int j = 0; j < 4; j++) {
            int r = (t >> 3) + (j << 5);
            int gm = m0 + r;
            if (gm < M) {
                if constexpr (ABF16) {
                    ra[j].u = *(const uint2*)&Ab[(size_t)gm * K + kb + c];
                } else {
                    float4 f = *(const float4*)&Af[(size_t)gm * K + kb + c];
                    ra[j].h[0] = (__bf16)f.x; ra[j].h[1] = (__bf16)f.y;
                    ra[j].h[2] = (__bf16)f.z; ra[j].h[3] = (__bf16)f.w;
                }
            } else { ra[j].u.x = 0u; ra[j].u.y = 0u; }
            int gn = n0 + r;
            if (gn < N) {
                float4 f = *(const float4*)&B[(size_t)gn * K + kb + c];
                rb[j].h[0] = (__bf16)f.x; rb[j].h[1] = (__bf16)f.y;
                rb[j].h[2] = (__bf16)f.z; rb[j].h[3] = (__bf16)f.w;
            } else { rb[j].u.x = 0u; rb[j].u.y = 0u; }
        }
    };
    auto store_tile = [&](pack4* ra, pack4* rb) {
#pragma unroll
        for (int j = 0; j < 4; j++) {
            int r = (t >> 3) + (j << 5);
            *(uint2*)&As[r * 32 + c] = ra[j].u;
            *(uint2*)&Bs[r * 32 + c] = rb[j].u;
        }
    };

    load_tile(0, pa, pb);
    store_tile(pa, pb);
    __syncthreads();

    for (int kb = 0; kb < K; kb += 32) {
        // prefetch next tile while computing current
        if (kb + 32 < K) load_tile(kb + 32, pa, pb);

        bf16x16 a0, a1;
        {
            int row = wm * 32 + lq;
            a0 = cat16(*(const bf16x8*)&As[row * 32 + hf * 8],
                       *(const bf16x8*)&As[row * 32 + 16 + hf * 8]);
            row += 16;
            a1 = cat16(*(const bf16x8*)&As[row * 32 + hf * 8],
                       *(const bf16x8*)&As[row * 32 + 16 + hf * 8]);
        }
        bf16x16 bfr[4];
#pragma unroll
        for (int nt = 0; nt < 4; nt++) {
            int n = wn * 64 + nt * 16 + lq;
            bfr[nt] = cat16(*(const bf16x8*)&Bs[n * 32 + hf * 8],
                            *(const bf16x8*)&Bs[n * 32 + 16 + hf * 8]);
        }
#pragma unroll
        for (int nt = 0; nt < 4; nt++) {
            acc[0][nt] = wmma_bf16(a0, bfr[nt], acc[0][nt]);
            acc[1][nt] = wmma_bf16(a1, bfr[nt], acc[1][nt]);
        }
        __syncthreads();
        if (kb + 32 < K) {
            store_tile(pa, pb);
            __syncthreads();
        }
    }

#pragma unroll
    for (int mi = 0; mi < 2; mi++)
#pragma unroll
        for (int nt = 0; nt < 4; nt++)
#pragma unroll
            for (int r = 0; r < 8; r++) {
                int gm = m0 + wm * 32 + mi * 16 + hf * 8 + r;
                int gn = n0 + wn * 64 + nt * 16 + lq;
                if (gm < M && gn < N) C[(size_t)gm * N + gn] = acc[mi][nt][r];
            }
}

// ---------------------------------------------------------------------------
// RMS norms (fp32 in -> bf16 out, rearranged head-major); V -> bf16 transposed
// ---------------------------------------------------------------------------
__global__ __launch_bounds__(128) void rmsnorm_q_kernel(const float* __restrict__ x,
                                                        const float* __restrict__ w,
                                                        __bf16* __restrict__ out, int Q) {
    const int q = blockIdx.x, h = blockIdx.y, tid = threadIdx.x;
    float v = x[(size_t)q * 4096 + h * 128 + tid];
    float ss = v * v;
#pragma unroll
    for (int off = 16; off > 0; off >>= 1) ss += __shfl_xor(ss, off, 32);
    __shared__ float wsum[4];
    if ((tid & 31) == 0) wsum[tid >> 5] = ss;
    __syncthreads();
    float tot = wsum[0] + wsum[1] + wsum[2] + wsum[3];
    float inv = rsqrtf(tot * (1.0f / 128.0f) + 1e-5f);
    out[((size_t)h * Q + q) * 128 + tid] = (__bf16)(v * inv * w[tid]);
}

__global__ __launch_bounds__(128) void rmsnorm_k_kernel(const float* __restrict__ x,
                                                        const float* __restrict__ w,
                                                        __bf16* __restrict__ out, int Kv) {
    const int kv = blockIdx.x, kvh = blockIdx.y, tid = threadIdx.x;
    float v = x[(size_t)kv * 1024 + kvh * 128 + tid];
    float ss = v * v;
#pragma unroll
    for (int off = 16; off > 0; off >>= 1) ss += __shfl_xor(ss, off, 32);
    __shared__ float wsum[4];
    if ((tid & 31) == 0) wsum[tid >> 5] = ss;
    __syncthreads();
    float tot = wsum[0] + wsum[1] + wsum[2] + wsum[3];
    float inv = rsqrtf(tot * (1.0f / 128.0f) + 1e-5f);
    out[((size_t)kvh * Kv + kv) * 128 + tid] = (__bf16)(v * inv * w[tid]);
}

// vf[kv][kvh*128+hd] (f32) -> vbt[(kvh*128+hd)][kv] (bf16, head-dim-major)
__global__ __launch_bounds__(256) void vtranspose_kernel(const float* __restrict__ x,
                                                         __bf16* __restrict__ out, int Kv) {
    size_t total = (size_t)Kv * 1024;
    for (size_t i = blockIdx.x * 256ull + threadIdx.x; i < total; i += gridDim.x * 256ull) {
        int kv = (int)(i >> 10);
        int cc = (int)(i & 1023);
        out[(size_t)cc * Kv + kv] = (__bf16)x[i];
    }
}

// ---------------------------------------------------------------------------
// Flash attention: block = (1 head, 128 q-rows), 8 waves x 16 q-rows.
// Q fragments in registers; K/V chunks staged by the Tensor Data Mover into
// double-buffered LDS; online softmax; WMMA QK^T and PV.
// ---------------------------------------------------------------------------
__global__ __launch_bounds__(256) void attn_kernel(const __bf16* __restrict__ qb,
                                                   const __bf16* __restrict__ kb,
                                                   const __bf16* __restrict__ vbt,
                                                   const float* __restrict__ mask,
                                                   __bf16* __restrict__ ob,
                                                   int Q, int Kv) {
    __shared__ __align__(16) __bf16 Ks[2][64 * 128];  // kv-major (row=kv, col=hd)
    __shared__ __align__(16) __bf16 Vt[2][128 * 64];  // hd-major (row=hd, col=kv)
    __shared__ __align__(16) __bf16 Ps[8 * 16 * 64];  // per-wave P strip

    const int t     = threadIdx.x;
    const int wave  = t >> 5, lane = t & 31;
    const int hf    = lane >> 4, lq = lane & 15;
    const int qtile = blockIdx.x, h = blockIdx.y;
    const int kvh   = h >> 2;  // GROUPS = 4
    const int qbase = qtile * 128;
    const float scale = 0.08838834764831845f;  // 1/sqrt(128)

    const unsigned int ldsK[2] = {
        (unsigned int)(uintptr_t)&Ks[0][0], (unsigned int)(uintptr_t)&Ks[1][0]};
    const unsigned int ldsV[2] = {
        (unsigned int)(uintptr_t)&Vt[0][0], (unsigned int)(uintptr_t)&Vt[1][0]};

    const int nchunks = (Kv + 63) >> 6;

    // Q fragments in registers for the full KV loop (4 x 32-K chunks)
    bf16x16 qfrag[4];
    {
        size_t row = (size_t)(h * Q + qbase + wave * 16 + lq) * 128;
#pragma unroll
        for (int kc = 0; kc < 4; kc++) {
            size_t base = row + kc * 32 + hf * 8;
            qfrag[kc] = cat16(*(const bf16x8*)&qb[base], *(const bf16x8*)&qb[base + 16]);
        }
    }

    float mrow[8], lrow[8];
    f32x8 o[8];
#pragma unroll
    for (int r = 0; r < 8; r++) { mrow[r] = -3.0e38f; lrow[r] = 0.f; }
#pragma unroll
    for (int n = 0; n < 8; n++) o[n] = (f32x8){0.f,0.f,0.f,0.f,0.f,0.f,0.f,0.f};

    // Prologue: TDM-load chunk 0 into buffer 0 (wave 0 drives the TDM).
    if (wave == 0) {
        int kv0 = 0;
        tdm_load_2d_bf16(ldsK[0], kb + ((size_t)kvh * Kv) * 128,
                         128u, (unsigned int)(Kv - kv0), 128u, 64u, 128ull);
        tdm_load_2d_bf16(ldsV[0], vbt + (size_t)kvh * 128 * Kv + kv0,
                         (unsigned int)(Kv - kv0), 128u, 64u, 128u,
                         (unsigned long long)Kv);
    }

    for (int ci = 0; ci < nchunks; ci++) {
        const int kv0 = ci << 6;
        const int buf = ci & 1;

        if (wave == 0) {
            if (ci + 1 < nchunks) {
                int kv1 = kv0 + 64;
                int nb = buf ^ 1;
                tdm_load_2d_bf16(ldsK[nb], kb + ((size_t)kvh * Kv + kv1) * 128,
                                 128u, (unsigned int)(Kv - kv1), 128u, 64u, 128ull);
                tdm_load_2d_bf16(ldsV[nb], vbt + (size_t)kvh * 128 * Kv + kv1,
                                 (unsigned int)(Kv - kv1), 128u, 64u, 128u,
                                 (unsigned long long)Kv);
                __builtin_amdgcn_s_wait_tensorcnt(2);  // chunk ci complete (in-order)
            } else {
                __builtin_amdgcn_s_wait_tensorcnt(0);
            }
        }
        __syncthreads();  // chunk ci visible to all waves

        const __bf16* Kc = &Ks[buf][0];
        const __bf16* Vc = &Vt[buf][0];

        // ---- S = Q * K^T for this wave's 16x64 strip ----
        f32x8 s[4];
#pragma unroll
        for (int nt = 0; nt < 4; nt++) s[nt] = (f32x8){0.f,0.f,0.f,0.f,0.f,0.f,0.f,0.f};
#pragma unroll
        for (int kc = 0; kc < 4; kc++) {
            bf16x16 bk[4];
#pragma unroll
            for (int nt = 0; nt < 4; nt++) {
                int n = nt * 16 + lq;
                bk[nt] = cat16(*(const bf16x8*)&Kc[n * 128 + kc * 32 + hf * 8],
                               *(const bf16x8*)&Kc[n * 128 + kc * 32 + hf * 8 + 16]);
            }
#pragma unroll
            for (int nt = 0; nt < 4; nt++) s[nt] = wmma_bf16(qfrag[kc], bk[nt], s[nt]);
        }

        // ---- scale + mask + online softmax (rows = hf*8 + r) ----
        float tmax[8];
#pragma unroll
        for (int r = 0; r < 8; r++) tmax[r] = -3.0e38f;
#pragma unroll
        for (int nt = 0; nt < 4; nt++) {
            int kvg = kv0 + nt * 16 + lq;
#pragma unroll
            for (int r = 0; r < 8; r++) {
                int qg = qbase + wave * 16 + hf * 8 + r;
                float val = (kvg < Kv)
                                ? s[nt][r] * scale + mask[(size_t)qg * Kv + kvg]
                                : -1.0e30f;
                s[nt][r] = val;
                tmax[r] = fmaxf(tmax[r], val);
            }
        }
#pragma unroll
        for (int r = 0; r < 8; r++) {
#pragma unroll
            for (int off = 1; off < 16; off <<= 1)
                tmax[r] = fmaxf(tmax[r], __shfl_xor(tmax[r], off, 32));
        }

        float mnew[8], alpha[8], psum[8];
#pragma unroll
        for (int r = 0; r < 8; r++) {
            mnew[r] = fmaxf(mrow[r], tmax[r]);
            alpha[r] = __expf(mrow[r] - mnew[r]);
            psum[r] = 0.f;
        }
#pragma unroll
        for (int nt = 0; nt < 4; nt++) {
#pragma unroll
            for (int r = 0; r < 8; r++) {
                float p = __expf(s[nt][r] - mnew[r]);
                psum[r] += p;
                Ps[wave * 1024 + (hf * 8 + r) * 64 + nt * 16 + lq] = (__bf16)p;
            }
        }
#pragma unroll
        for (int r = 0; r < 8; r++) {
#pragma unroll
            for (int off = 1; off < 16; off <<= 1) psum[r] += __shfl_xor(psum[r], off, 32);
            lrow[r] = lrow[r] * alpha[r] + psum[r];
            mrow[r] = mnew[r];
        }
#pragma unroll
        for (int n = 0; n < 8; n++)
#pragma unroll
            for (int r = 0; r < 8; r++) o[n][r] *= alpha[r];

        asm volatile("s_wait_dscnt 0" ::: "memory");  // P writes -> P reads (same wave)

        // ---- O += P * V  (P: 16x64 A-operand from LDS, V^T gives contiguous B) ----
#pragma unroll
        for (int kc2 = 0; kc2 < 2; kc2++) {
            int pbase = wave * 1024 + lq * 64 + kc2 * 32 + hf * 8;
            bf16x16 ap = cat16(*(const bf16x8*)&Ps[pbase], *(const bf16x8*)&Ps[pbase + 16]);
#pragma unroll
            for (int nt2 = 0; nt2 < 8; nt2++) {
                int n = nt2 * 16 + lq;
                bf16x16 bv = cat16(*(const bf16x8*)&Vc[n * 64 + kc2 * 32 + hf * 8],
                                   *(const bf16x8*)&Vc[n * 64 + kc2 * 32 + hf * 8 + 16]);
                o[nt2] = wmma_bf16(ap, bv, o[nt2]);
            }
        }
        __syncthreads();  // compute done before this buffer is overwritten
    }

    // ---- normalize and store to [Q][H*HD] bf16 ----
#pragma unroll
    for (int nt2 = 0; nt2 < 8; nt2++)
#pragma unroll
        for (int r = 0; r < 8; r++) {
            int qg = qbase + wave * 16 + hf * 8 + r;
            int hd = nt2 * 16 + lq;
            float val = o[nt2][r] / lrow[r];
            if (qg < Q) ob[(size_t)qg * 4096 + h * 128 + hd] = (__bf16)val;
        }
}

// ---------------------------------------------------------------------------
extern "C" void kernel_launch(void* const* d_in, const int* in_sizes, int n_in,
                              void* d_out, int out_size, void* d_ws, size_t ws_size,
                              hipStream_t stream) {
    const float* hidden = (const float*)d_in[0];
    const float* cross  = (const float*)d_in[1];
    const float* mask   = (const float*)d_in[2];
    const float* Wq     = (const float*)d_in[3];
    const float* Wk     = (const float*)d_in[4];
    const float* Wv     = (const float*)d_in[5];
    const float* Wo     = (const float*)d_in[6];
    const float* qw     = (const float*)d_in[7];
    const float* kw     = (const float*)d_in[8];

    const int D = 4096, H = 32, KVH = 8, HD = 128;
    const int Q  = in_sizes[0] / D;   // 1024
    const int Kv = in_sizes[1] / D;   // 6404
    const int NKV = KVH * HD;         // 1024

    char* ws = (char*)d_ws;
    size_t off = 0;
    float*  qf  = (float*)(ws + off);  off += (size_t)Q * D * 4;
    float*  kf  = (float*)(ws + off);  off += (size_t)Kv * NKV * 4;
    float*  vf  = (float*)(ws + off);  off += (size_t)Kv * NKV * 4;
    __bf16* qb  = (__bf16*)(ws + off); off += (size_t)Q * D * 2;
    __bf16* kb  = (__bf16*)(ws + off); off += (size_t)Kv * NKV * 2;
    __bf16* vbt = (__bf16*)(ws + off); off += (size_t)Kv * NKV * 2;
    __bf16* ab  = (__bf16*)(ws + off); off += (size_t)Q * D * 2;

    dim3 blk(256);
    // projections
    gemm_nt_kernel<false><<<dim3(D / 128, (Q + 127) / 128), blk, 0, stream>>>(hidden, Wq, qf, Q, D, D);
    gemm_nt_kernel<false><<<dim3(NKV / 128, (Kv + 127) / 128), blk, 0, stream>>>(cross, Wk, kf, Kv, NKV, D);
    gemm_nt_kernel<false><<<dim3(NKV / 128, (Kv + 127) / 128), blk, 0, stream>>>(cross, Wv, vf, Kv, NKV, D);
    // norms / converts (rearrange to head-major bf16; V transposed for TDM)
    rmsnorm_q_kernel<<<dim3(Q, H), 128, 0, stream>>>(qf, qw, qb, Q);
    rmsnorm_k_kernel<<<dim3(Kv, KVH), 128, 0, stream>>>(kf, kw, kb, Kv);
    vtranspose_kernel<<<dim3(4096), blk, 0, stream>>>(vf, vbt, Kv);
    // flash attention (TDM-staged KV, double-buffered)
    attn_kernel<<<dim3(Q / 128, H), blk, 0, stream>>>(qb, kb, vbt, mask, ab, Q, Kv);
    // output projection -> f32 d_out
    gemm_nt_kernel<true><<<dim3(D / 128, (Q + 127) / 128), blk, 0, stream>>>(ab, Wo, (float*)d_out, Q, D, D);
}
